// DGCRM_88227218194820
// MI455X (gfx1250) — compile-verified, matrix operand order checked by reading, not run on hardware
//
#include <hip/hip_runtime.h>
#include <hip/hip_bf16.h>

// ---------------------------------------------------------------------------
// DGCRM on gfx1250 (MI455X).
// Live dataflow only (the hyper/DGCN branch is dead code in the reference):
//   z  = sigmoid([x,h] @ Wz + bz)          Wz folded from rnn_W[0]+rnn_W[1]
//   r  = sigmoid([x,h] @ Wr + br)          Wr folded from rnn_W[2]+rnn_W[3]
//   hc = tanh   ([x,r*h] @ Wc + bc)        Wc folded from rnn_W[4]+rnn_W[5]
//   out = z*h + (1-z)*hc
// 16384 x 66 -> 64 projections as v_wmma_f32_16x16x32_f16 (K=64 hidden part,
// two K-steps of 32) + VALU FMA for the K=2 x-part folded into C-init.
// Data movement: GLOBAL_LOAD_ASYNC_TO_LDS_B128 stages the hidden tile,
// GLOBAL_STORE_ASYNC_FROM_LDS_B128 drains the coalesced output (ASYNCcnt).
// ---------------------------------------------------------------------------

typedef __attribute__((ext_vector_type(16))) _Float16 v16h;
typedef __attribute__((ext_vector_type(8)))  float    v8f;

#define M_ROWS   16384          // B*N = 16*1024
#define GC_ALPHA 0.05f

// workspace layout (bytes)
#define WS_BOPS_OFF 0           // _Float16 [3 gates][4 ntiles][2 ksteps][32 lanes][16] = 12288 halves
#define WS_WX_OFF   24576       // float [3][2][64]   x-part weights
#define WS_BEFF_OFF 26112       // float [3][64]      fused biases

// ---------------------------------------------------------------------------
// Prep: fold rnn_W pairs (with the 0.05-scaled concat copies) into effective
// 66x64 weights; hidden part (rows 2..65) pre-swizzled into WMMA B-operand
// layout: lane l holds column n = (l&15), half hi = l>>4; element e maps to
// K = (e>>3)*16 + hi*8 + (e&7)   (ISA 7.12.2, 16-bit operand table).
// ---------------------------------------------------------------------------
__global__ void dgcrm_prep(const float* __restrict__ W, const float* __restrict__ bias,
                           _Float16* __restrict__ Bops, float* __restrict__ Wx,
                           float* __restrict__ beff)
{
    int idx = blockIdx.x * blockDim.x + threadIdx.x;
    if (idx < 12288) {
        int g = idx >> 12;            // gate 0..2
        int rem = idx & 4095;
        int k = rem >> 6;             // 0..63 (hidden K)
        int n = rem & 63;             // 0..63 (output col)
        int i0 = 2 * g, i1 = 2 * g + 1;
        auto wr = [&](int i, int row) { return W[(i * 198 + row) * 64 + n]; };
        float we = wr(i0, 2 + k) + GC_ALPHA * (wr(i0, 68 + k) + wr(i0, 134 + k))
                 + wr(i1, 2 + k) + GC_ALPHA * (wr(i1, 68 + k) + wr(i1, 134 + k));
        int s    = k >> 5;            // k-step (0/1)
        int krel = k & 31;
        int hi   = (krel >> 3) & 1;
        int e    = ((krel >> 4) << 3) | (krel & 7);
        int lane = (hi << 4) | (n & 15);
        int t    = n >> 4;            // n-tile
        Bops[((((g * 4 + t) * 2 + s) << 5) + lane) * 16 + e] = (_Float16)we;
    } else if (idx < 12288 + 384) {
        int j = idx - 12288;
        int g = j >> 7, k = (j >> 6) & 1, n = j & 63;
        int i0 = 2 * g, i1 = 2 * g + 1;
        auto wr = [&](int i, int row) { return W[(i * 198 + row) * 64 + n]; };
        Wx[j] = wr(i0, k) + GC_ALPHA * (wr(i0, 66 + k) + wr(i0, 132 + k))
              + wr(i1, k) + GC_ALPHA * (wr(i1, 66 + k) + wr(i1, 132 + k));
    } else if (idx < 12288 + 384 + 192) {
        int j = idx - 12288 - 384;
        int g = j >> 6, n = j & 63;
        beff[j] = bias[(2 * g) * 64 + n] + bias[(2 * g + 1) * 64 + n];
    }
}

__device__ __forceinline__ float fast_sigmoid(float v) {
    return 1.0f / (1.0f + __expf(-v));          // v_exp_f32 (TRANS, co-exec w/ WMMA)
}
__device__ __forceinline__ float fast_tanh(float v) {
    return 2.0f / (1.0f + __expf(-2.0f * v)) - 1.0f;
}

// Generic pointers to LDS carry the LDS byte offset in their low 32 bits
__device__ __forceinline__ unsigned lds_off(const void* p) {
    return (unsigned)(unsigned long long)p;
}

// ---------------------------------------------------------------------------
// Main fused GRU kernel: 1 wave == one 16-row tile of M=16384.
// 256 blocks x 128 threads (4 waves). LDS is wave-private -> no barriers.
// ---------------------------------------------------------------------------
__global__ void __launch_bounds__(128) dgcrm_main(
    const float* __restrict__ x, const float* __restrict__ hid,
    const _Float16* __restrict__ Bops, const float* __restrict__ Wx,
    const float* __restrict__ beff, float* __restrict__ out)
{
    __shared__ float lds_h[4][16][68];   // hidden tile, stride 68 vs bank conflicts
    __shared__ float lds_r[4][16][68];   // r gate re-swizzle, then output staging

    const int lane = threadIdx.x & 31;
    const int w    = threadIdx.x >> 5;
    const int r0   = (blockIdx.x * 4 + w) * 16;   // first row of this tile

    __builtin_prefetch(Bops, 0, 3);               // weights -> global_prefetch

    // --- stage hidden tile (16 x 64 f32) global -> LDS via async DMA path ---
    const float4* h4 = (const float4*)(hid + r0 * 64);
#pragma unroll
    for (int i = 0; i < 8; ++i) {
        int q = i * 32 + lane;                    // float4 index in tile (0..255)
        unsigned long long ga = (unsigned long long)(h4 + q);
        unsigned la = lds_off(&lds_h[w][q >> 4][(q & 15) * 4]);
        asm volatile("global_load_async_to_lds_b128 %0, %1, off"
                     :: "v"(la), "v"(ga) : "memory");
    }
    asm volatile("s_wait_asynccnt 0x0" ::: "memory");

    const int m   = lane & 15;                    // A row / C col
    const int hi8 = (lane >> 4) * 8;              // lane-half K/M offset

    // A operands: hidden tile in 16-bit 16x32 A layout (ISA 7.12.2)
    v16h a0, a1;
#pragma unroll
    for (int e = 0; e < 8; ++e) {
        a0[e]     = (_Float16)lds_h[w][m][hi8 + e];
        a0[e + 8] = (_Float16)lds_h[w][m][16 + hi8 + e];
        a1[e]     = (_Float16)lds_h[w][m][32 + hi8 + e];
        a1[e + 8] = (_Float16)lds_h[w][m][48 + hi8 + e];
    }

    // x values for the K=2 part of the C-init (C layout: row hi8+j, col m)
    float2 xv[8];
#pragma unroll
    for (int j = 0; j < 8; ++j)
        xv[j] = ((const float2*)x)[r0 + hi8 + j];

    const v16h* Bv = (const v16h*)Bops;           // per-lane B operands

    float zf[4][8];
#pragma unroll
    for (int t = 0; t < 4; ++t) {
        const int n = t * 16 + m;
        v8f cz, cr;
#pragma unroll
        for (int j = 0; j < 8; ++j) {
            cz[j] = beff[n]      + xv[j].x * Wx[n]       + xv[j].y * Wx[64 + n];
            cr[j] = beff[64 + n] + xv[j].x * Wx[128 + n] + xv[j].y * Wx[192 + n];
        }
        cz = __builtin_amdgcn_wmma_f32_16x16x32_f16(false, a0, false, Bv[((0 * 4 + t) * 2 + 0) * 32 + lane], (short)0, cz, false, false);
        cz = __builtin_amdgcn_wmma_f32_16x16x32_f16(false, a1, false, Bv[((0 * 4 + t) * 2 + 1) * 32 + lane], (short)0, cz, false, false);
        cr = __builtin_amdgcn_wmma_f32_16x16x32_f16(false, a0, false, Bv[((1 * 4 + t) * 2 + 0) * 32 + lane], (short)0, cr, false, false);
        cr = __builtin_amdgcn_wmma_f32_16x16x32_f16(false, a1, false, Bv[((1 * 4 + t) * 2 + 1) * 32 + lane], (short)0, cr, false, false);
#pragma unroll
        for (int j = 0; j < 8; ++j) {
            zf[t][j] = fast_sigmoid(cz[j]);
            lds_r[w][hi8 + j][n] = fast_sigmoid(cr[j]);   // park r for re-swizzle
        }
    }

    // candidate A operands: (r * hidden) in A layout, f16
    v16h ca0, ca1;
#pragma unroll
    for (int e = 0; e < 8; ++e) {
        ca0[e]     = (_Float16)(lds_r[w][m][hi8 + e]      * lds_h[w][m][hi8 + e]);
        ca0[e + 8] = (_Float16)(lds_r[w][m][16 + hi8 + e] * lds_h[w][m][16 + hi8 + e]);
        ca1[e]     = (_Float16)(lds_r[w][m][32 + hi8 + e] * lds_h[w][m][32 + hi8 + e]);
        ca1[e + 8] = (_Float16)(lds_r[w][m][48 + hi8 + e] * lds_h[w][m][48 + hi8 + e]);
    }

    // hc + gating; stage result into lds_r (r already consumed) for coalesced
    // async store-from-LDS.
#pragma unroll
    for (int t = 0; t < 4; ++t) {
        const int n = t * 16 + m;
        v8f cc;
#pragma unroll
        for (int j = 0; j < 8; ++j)
            cc[j] = beff[128 + n] + xv[j].x * Wx[256 + n] + xv[j].y * Wx[320 + n];
        cc = __builtin_amdgcn_wmma_f32_16x16x32_f16(false, ca0, false, Bv[((2 * 4 + t) * 2 + 0) * 32 + lane], (short)0, cc, false, false);
        cc = __builtin_amdgcn_wmma_f32_16x16x32_f16(false, ca1, false, Bv[((2 * 4 + t) * 2 + 1) * 32 + lane], (short)0, cc, false, false);
#pragma unroll
        for (int j = 0; j < 8; ++j) {
            float hc = fast_tanh(cc[j]);
            float ho = lds_h[w][hi8 + j][n];
            float z  = zf[t][j];
            lds_r[w][hi8 + j][n] = z * ho + (1.0f - z) * hc;
        }
    }

    // DS writes must land before the async engine reads LDS (separate counters)
    asm volatile("s_wait_dscnt 0x0" ::: "memory");

    float4* o4 = (float4*)(out + r0 * 64);
#pragma unroll
    for (int i = 0; i < 8; ++i) {
        int q = i * 32 + lane;
        unsigned long long ga = (unsigned long long)(o4 + q);
        unsigned la = lds_off(&lds_r[w][q >> 4][(q & 15) * 4]);
        asm volatile("global_store_async_from_lds_b128 %0, %1, off"
                     :: "v"(ga), "v"(la) : "memory");
    }
    // async stores drain on ASYNCcnt; S_ENDPGM performs an implicit wait-idle.
}

// ---------------------------------------------------------------------------
extern "C" void kernel_launch(void* const* d_in, const int* in_sizes, int n_in,
                              void* d_out, int out_size, void* d_ws, size_t ws_size,
                              hipStream_t stream) {
    (void)in_sizes; (void)n_in; (void)out_size; (void)ws_size;
    const float* x    = (const float*)d_in[0];   // [16,1024,2]
    const float* hid  = (const float*)d_in[1];   // [16,1024,64]
    const float* rnnW = (const float*)d_in[12];  // [6,198,64]
    const float* rnnb = (const float*)d_in[13];  // [6,64]

    _Float16* Bops = (_Float16*)((char*)d_ws + WS_BOPS_OFF);
    float*    Wx   = (float*)((char*)d_ws + WS_WX_OFF);
    float*    beff = (float*)((char*)d_ws + WS_BEFF_OFF);

    dgcrm_prep<<<51, 256, 0, stream>>>(rnnW, rnnb, Bops, Wx, beff);
    dgcrm_main<<<256, 128, 0, stream>>>(x, hid, Bops, Wx, beff, (float*)d_out);
}
